// SelfAttention_43181601194274
// MI455X (gfx1250) — matrix operand order repaired and checked by reading, not compiled, not run on hardware
//
#include <hip/hip_runtime.h>

// CDNA5 (gfx1250) fused flash-attention, bf16 WMMA, f32 accumulate.
// v_wmma_f32_16x16x32_bf16 (wave32), ping-pong LDS tiles, pipelined B-fragment loads.

typedef __attribute__((ext_vector_type(16))) __bf16 v16bf;
typedef __attribute__((ext_vector_type(8)))  float  v8f;

#define BB   16
#define NN   2048
#define DD   256
#define BQ   128          // queries per workgroup (8 waves x 16 rows)
#define BK   64           // keys per tile
#define NT   (NN / BK)    // 32 key tiles
#define KSTR 264          // bf16 stride of K rows in LDS (256 + 8 pad -> conflict-free b128)
#define VSTR 72           // bf16 stride of V^T rows (64 + 8 pad)
#define PSTR 72           // bf16 stride of P staging rows
#define LOG2E 1.4426950408889634f
#define NEGV  -2.0e20f

struct SmemT {
  __align__(16) __bf16 k[2][BK * KSTR];   // x2 tile, row-major [key][d], ping-pong
  __align__(16) __bf16 vt[2][DD * VSTR];  // x2 tile, transposed [d][key], ping-pong
  __align__(16) __bf16 p[8 * 16 * PSTR];  // per-wave P staging [row][key]
  unsigned char mask[2][BK];
};

// 16 contiguous bf16 (unused; B-fragments are loaded with ld16_split)
__device__ inline v16bf ld16(const __bf16* p) {
  union { uint4 u[2]; v16bf v; } t;
  t.u[0] = *(const uint4*)(p);
  t.u[1] = *(const uint4*)(p + 8);
  return t.v;
}
// A-fragment per-lane data: elements 0..7 at +0, elements 8..15 at +16 (K base+16..23)
__device__ inline v16bf ld16_split(const __bf16* p) {
  union { uint4 u[2]; v16bf v; } t;
  t.u[0] = *(const uint4*)(p);
  t.u[1] = *(const uint4*)(p + 16);
  return t.v;
}

// Stage one 64x256 f32 tile of x2 into LDS as bf16: row-major copy + d-major transpose.
__device__ inline void stage_tile(const float* __restrict__ kg,
                                  const unsigned char* __restrict__ mg,
                                  __bf16* kdst, __bf16* vdst, unsigned char* mdst,
                                  int tid) {
  #pragma unroll
  for (int it = 0; it < 16; ++it) {            // 4096 float4, coalesced along d
    int idx = it * 256 + tid;
    int row = idx >> 6, c4 = idx & 63;
    float4 f = *(const float4*)(kg + row * DD + c4 * 4);
    union { __bf16 h[4]; uint2 u; } pk;
    pk.h[0]=(__bf16)f.x; pk.h[1]=(__bf16)f.y; pk.h[2]=(__bf16)f.z; pk.h[3]=(__bf16)f.w;
    *(uint2*)(&kdst[row * KSTR + c4 * 4]) = pk.u;
  }
  #pragma unroll
  for (int it = 0; it < 16; ++it) {            // k-fast mapping: contiguous LDS stores
    int idx = it * 256 + tid;
    int k = idx & 63, d4 = idx >> 6;
    float4 f = *(const float4*)(kg + k * DD + d4 * 4);
    vdst[(d4*4 + 0) * VSTR + k] = (__bf16)f.x;
    vdst[(d4*4 + 1) * VSTR + k] = (__bf16)f.y;
    vdst[(d4*4 + 2) * VSTR + k] = (__bf16)f.z;
    vdst[(d4*4 + 3) * VSTR + k] = (__bf16)f.w;
  }
  if (tid < BK) mdst[tid] = mg[tid];
}

__global__ __launch_bounds__(256)
void attn_fa_bf16_wmma(const float* __restrict__ x1, const float* __restrict__ x2,
                       const unsigned char* __restrict__ x2_mask,
                       float* __restrict__ out) {
  __shared__ SmemT sm;
  const int tid  = threadIdx.x;
  const int lane = tid & 31;
  const int wave = tid >> 5;
  const int col  = lane & 15;          // N index inside 16x16 tile / A row
  const int half = (lane >> 4) & 1;    // lane group 0..15 vs 16..31
  const int b    = blockIdx.y;
  const int q0   = blockIdx.x * BQ;
  const int qg_base = q0 + wave * 16;  // global query row of this wave's row 0

  const float*         x2b = x2 + (size_t)b * NN * DD;
  const unsigned char* mkb = x2_mask + (size_t)b * NN;

  // ---------------- Q -> 8 bf16 A-fragments (16x32 each, d = kk*32..+31) ----
  v16bf qf[8];
  {
    const float* qp = x1 + ((size_t)b * NN + (qg_base + col)) * DD;
    #pragma unroll
    for (int kk = 0; kk < 8; ++kk) {
      const int d0 = kk * 32 + (half ? 8 : 0);
      float4 a0 = *(const float4*)(qp + d0);
      float4 a1 = *(const float4*)(qp + d0 + 4);
      float4 a2 = *(const float4*)(qp + d0 + 16);
      float4 a3 = *(const float4*)(qp + d0 + 20);
      float f[16] = {a0.x,a0.y,a0.z,a0.w, a1.x,a1.y,a1.z,a1.w,
                     a2.x,a2.y,a2.z,a2.w, a3.x,a3.y,a3.z,a3.w};
      v16bf v;
      #pragma unroll
      for (int i = 0; i < 16; ++i) v[i] = (__bf16)f[i];
      qf[kk] = v;
    }
  }

  // ---------------- accumulators / online-softmax state ---------------------
  v8f Oacc[16];
  #pragma unroll
  for (int dt = 0; dt < 16; ++dt) Oacc[dt] = (v8f){};
  float m_i[8], l_i[8];
  #pragma unroll
  for (int r = 0; r < 8; ++r) { m_i[r] = -3.0e38f; l_i[r] = 0.0f; }

  // prologue: stage tile 0 into buffer 0
  stage_tile(x2b, mkb, sm.k[0], sm.vt[0], sm.mask[0], tid);
  __syncthreads();

  // ---------------- key tile loop (one barrier per iteration) ---------------
  for (int i = 0; i < NT; ++i) {
    const int kb  = i * BK;
    const int buf = i & 1;

    // stage tile i+1 into the other buffer while computing on this one
    if (i + 1 < NT) {
      if (i + 2 < NT)   // speculative prefetch two tiles ahead (64KB, 256B/thread)
        __builtin_prefetch(x2b + (size_t)(kb + 2 * BK) * DD + tid * 64, 0, 1);
      stage_tile(x2b + (size_t)(kb + BK) * DD, mkb + kb + BK,
                 sm.k[buf ^ 1], sm.vt[buf ^ 1], sm.mask[buf ^ 1], tid);
    }

    const __bf16* curk = sm.k[buf];
    const __bf16* curv = sm.vt[buf];

    // ---- S = Q * K^T : 32 WMMA, B-fragments pipelined through 2 reg buffers
    v8f S[4];
    {
      const __bf16* kp = curk + col * KSTR + (half ? 8 : 0);
      v16bf bk = ld16_split(kp);              // step 0: t=0, kk=0
      #pragma unroll
      for (int t = 0; t < 4; ++t) {
        v8f acc = (v8f){};
        #pragma unroll
        for (int kk = 0; kk < 8; ++kk) {
          const int s = t * 8 + kk;
          v16bf bnext = bk;
          if (s < 31) {
            const int sn = s + 1;
            bnext = ld16_split(kp + (sn >> 3) * (16 * KSTR) + (sn & 7) * 32);
          }
          acc = __builtin_amdgcn_wmma_f32_16x16x32_bf16(false, qf[kk], false, bk,
                                                        (short)0, acc, false, false);
          bk = bnext;
        }
        S[t] = acc;
      }
    }

    // ---- masks: key padding fill (-2e20), then diagonal multiplied to 0 ----
    #pragma unroll
    for (int t = 0; t < 4; ++t) {
      const int  key = kb + t * 16 + col;
      const bool km  = sm.mask[buf][t * 16 + col] != 0;
      #pragma unroll
      for (int r = 0; r < 8; ++r) {
        float s = S[t][r];
        if (km) s = NEGV;
        if ((qg_base + r + 8 * half) == key) s = 0.0f;
        S[t][r] = s;
      }
    }

    // ---- online softmax (rows = VGPR idx + 8*half; reduce over 16-lane halves)
    float tm[8];
    #pragma unroll
    for (int r = 0; r < 8; ++r)
      tm[r] = fmaxf(fmaxf(S[0][r], S[1][r]), fmaxf(S[2][r], S[3][r]));
    #pragma unroll
    for (int xm = 1; xm <= 8; xm <<= 1) {
      #pragma unroll
      for (int r = 0; r < 8; ++r) tm[r] = fmaxf(tm[r], __shfl_xor(tm[r], xm, 32));
    }
    float sc[8];
    #pragma unroll
    for (int r = 0; r < 8; ++r) {
      float nm = fmaxf(m_i[r], tm[r]);
      sc[r] = __builtin_amdgcn_exp2f((m_i[r] - nm) * LOG2E);
      m_i[r] = nm;
    }
    float rs[8];
    #pragma unroll
    for (int r = 0; r < 8; ++r) rs[r] = 0.0f;
    #pragma unroll
    for (int t = 0; t < 4; ++t) {
      #pragma unroll
      for (int r = 0; r < 8; ++r) {
        float p = __builtin_amdgcn_exp2f((S[t][r] - m_i[r]) * LOG2E);
        S[t][r] = p;
        rs[r] += p;
      }
    }
    #pragma unroll
    for (int xm = 1; xm <= 8; xm <<= 1) {
      #pragma unroll
      for (int r = 0; r < 8; ++r) rs[r] += __shfl_xor(rs[r], xm, 32);
    }
    #pragma unroll
    for (int r = 0; r < 8; ++r) l_i[r] = l_i[r] * sc[r] + rs[r];
    #pragma unroll
    for (int dt = 0; dt < 16; ++dt) {
      #pragma unroll
      for (int r = 0; r < 8; ++r) Oacc[dt][r] *= sc[r];
    }

    // ---- P: C-layout -> A-layout via wave-private LDS staging --------------
    __bf16* myP = &sm.p[wave * 16 * PSTR];
    #pragma unroll
    for (int t = 0; t < 4; ++t) {
      #pragma unroll
      for (int r = 0; r < 8; ++r)
        myP[(r + 8 * half) * PSTR + t * 16 + col] = (__bf16)S[t][r];
    }
    asm volatile("s_wait_dscnt 0" ::: "memory");  // wave-local RAW fence
    v16bf pf[2];
    #pragma unroll
    for (int c = 0; c < 2; ++c)
      pf[c] = ld16_split(&myP[col * PSTR + c * 32 + 8 * half]);

    // ---- O += P * V : 32 WMMA, B-fragments pipelined -----------------------
    {
      const __bf16* vp = curv + col * VSTR + (half ? 8 : 0);
      v16bf bv = ld16_split(vp);              // step 0: c=0, dt=0
      #pragma unroll
      for (int c = 0; c < 2; ++c) {
        #pragma unroll
        for (int dt = 0; dt < 16; ++dt) {
          const int s = c * 16 + dt;
          v16bf bnext = bv;
          if (s < 31) {
            const int sn = s + 1;
            bnext = ld16_split(vp + (sn & 15) * (16 * VSTR) + (sn >> 4) * 32);
          }
          Oacc[dt] = __builtin_amdgcn_wmma_f32_16x16x32_bf16(false, pf[c], false, bv,
                                                             (short)0, Oacc[dt],
                                                             false, false);
          bv = bnext;
        }
      }
    }

    __syncthreads();  // staged tile i+1 visible; everyone done with buffer `buf`
  }

  // ---------------- epilogue: normalize and store f32 -----------------------
  float inv[8];
  #pragma unroll
  for (int r = 0; r < 8; ++r) inv[r] = __builtin_amdgcn_rcpf(l_i[r]);
  float* op = out + ((size_t)b * NN + qg_base) * DD;
  #pragma unroll
  for (int dt = 0; dt < 16; ++dt) {
    #pragma unroll
    for (int r = 0; r < 8; ++r)
      op[(r + 8 * half) * DD + dt * 16 + col] = Oacc[dt][r] * inv[r];
  }
}

extern "C" void kernel_launch(void* const* d_in, const int* in_sizes, int n_in,
                              void* d_out, int out_size, void* d_ws, size_t ws_size,
                              hipStream_t stream) {
  (void)in_sizes; (void)n_in; (void)out_size; (void)d_ws; (void)ws_size;
  const float*         x1 = (const float*)d_in[0];
  const float*         x2 = (const float*)d_in[1];
  const unsigned char* mk = (const unsigned char*)d_in[2];  // jnp.bool_ = 1 byte
  float*              out = (float*)d_out;
  dim3 grid(NN / BQ, BB);   // (16 query blocks, 16 batches)
  dim3 block(256);          // 8 wave32 per workgroup
  hipLaunchKernelGGL(attn_fa_bf16_wmma, grid, block, 0, stream, x1, x2, mk, out);
}
